// encoder_fuser_695784702020
// MI455X (gfx1250) — matrix operand order repaired and checked by reading, not compiled, not run on hardware
//
#include <hip/hip_runtime.h>
#include <math.h>

typedef __attribute__((ext_vector_type(16))) _Float16 v16h;
typedef __attribute__((ext_vector_type(8)))  _Float16 v8h;
typedef __attribute__((ext_vector_type(8)))  float    v8f;
typedef _Float16 half_t;

#define DMODEL 512
#define DFF    2048
#define NHEAD  8
#define HDIM   64
#define TSEQ   513          // 512 frames + 1 token
#define MPAD   640          // 5 * 128 row tiles
#define SPAD   544          // attention K-dim pad (17 * 32)
#define BATCH  2
#define SCALE_Q 0.125f      // 64^-0.5

// ---------------------------------------------------------------------------
// Generic batched GEMM:  C[z] = act( alpha * A[z] x B[z]^T + bias + resid )
// A: f16 [M x K], row-major, k-contiguous, row stride lda.
// B: f16 [N x K], row-major, k-contiguous, row stride ldbn (weight layout).
// z decomposed as h = z % zdiv, b = z / zdiv with per-component strides.
// Tile 128x64x32, double-buffered LDS; 256 threads = 8 waves; each wave owns a
// 32x32 sub-tile -> 4 x v_wmma_f32_16x16x32_f16 per K-step (2 A-frags reused
// across 2 B-frags), one barrier per K-step.
// transCh=1 stores the f16 result transposed: Ch[col*ldch + row]  (for V^T).
// ---------------------------------------------------------------------------
__global__ __launch_bounds__(256)
void gemm_wmma_f16(const half_t* __restrict__ A, long long sAh, long long sAb, int lda,
                   const half_t* __restrict__ B, long long sBh, long long sBb, int ldbn,
                   int M, int N, int K, int zdiv,
                   const float* __restrict__ bias,
                   const float* __restrict__ resid, long long sRh, long long sRb, int ldr,
                   float* __restrict__ Cf, long long sCfh, long long sCfb, int ldcf,
                   half_t* __restrict__ Ch, long long sChh, long long sChb, int ldch,
                   int transCh, float alpha, int act)
{
    __shared__ half_t As[2][128 * 32];   // 8 KB x2
    __shared__ half_t Bs[2][64 * 32];    // 4 KB x2

    const int z = blockIdx.z;
    const int h = z % zdiv;
    const int b = z / zdiv;
    A += (long long)h * sAh + (long long)b * sAb;
    B += (long long)h * sBh + (long long)b * sBb;
    if (resid) resid += (long long)h * sRh + (long long)b * sRb;
    if (Cf)    Cf    += (long long)h * sCfh + (long long)b * sCfb;
    if (Ch)    Ch    += (long long)h * sChh + (long long)b * sChb;

    const int tid  = threadIdx.x;
    const int lane = tid & 31;
    const int wave = tid >> 5;
    const int wmw  = wave >> 1;      // 0..3 : 32-row slice of the 128-row tile
    const int wnw  = wave & 1;       // 0..1 : 32-col slice of the 64-col tile
    const int m0   = blockIdx.y * 128;
    const int n0   = blockIdx.x * 64;

    // staging decomposition
    const int arow = tid >> 1;            // 0..127
    const int akc  = (tid & 1) * 16;      // 0,16   (two v8h each)
    const int brow = tid >> 2;            // 0..63
    const int bkc  = (tid & 3) * 8;       // 0,8,16,24

    v8f acc[2][2] = {{{}, {}}, {{}, {}}};

    auto stage = [&](int bsel, int kk) {
        const half_t* asrc = A + (long long)(m0 + arow) * lda + (kk + akc);
        *(v8h*)&As[bsel][arow * 32 + akc]     = *(const v8h*)asrc;
        *(v8h*)&As[bsel][arow * 32 + akc + 8] = *(const v8h*)(asrc + 8);
        if (kk + 32 < K)
            __builtin_prefetch(asrc + 32, 0, 1);     // global_prefetch next slab
        const half_t* bsrc = B + (long long)(n0 + brow) * ldbn + (kk + bkc);
        *(v8h*)&Bs[bsel][brow * 32 + bkc] = *(const v8h*)bsrc;
    };

    stage(0, 0);
    __syncthreads();

    const int fr  = lane & 15;
    const int sel = lane >> 4;

    int buf = 0;
    for (int k0 = 0; k0 < K; k0 += 32) {
        if (k0 + 32 < K) stage(buf ^ 1, k0 + 32);

        // A fragments: row = wmw*32 + i*16 + fr;
        // lanes 0-15 hold K{0..15}, lanes 16-31 hold K{16..31}
        v16h af[2];
        for (int i = 0; i < 2; ++i) {
            const half_t* ap = &As[buf][(wmw * 32 + i * 16 + fr) * 32 + sel * 16];
            union { v8h h8[2]; v16h h16; } ua;
            ua.h8[0] = *(const v8h*)ap;
            ua.h8[1] = *(const v8h*)(ap + 8);
            af[i] = ua.h16;
        }
        // B fragments: col = wnw*32 + j*16 + fr; 16 contiguous K from sel*16
        v16h bf[2];
        for (int j = 0; j < 2; ++j)
            bf[j] = *(const v16h*)&Bs[buf][(wnw * 32 + j * 16 + fr) * 32 + sel * 16];

        for (int i = 0; i < 2; ++i)
            for (int j = 0; j < 2; ++j)
                acc[i][j] = __builtin_amdgcn_wmma_f32_16x16x32_f16(
                    false, af[i], false, bf[j], (short)0, acc[i][j], false, false);

        __syncthreads();
        buf ^= 1;
    }

    // epilogue: C element (VGPR r, lane) -> row r + 8*(lane>>4), col lane&15
    for (int i = 0; i < 2; ++i) {
        for (int j = 0; j < 2; ++j) {
            const int col = n0 + wnw * 32 + j * 16 + fr;
            if (col >= N) continue;
            const float bv = bias ? bias[col] : 0.0f;
            for (int r = 0; r < 8; ++r) {
                const int row = m0 + wmw * 32 + i * 16 + sel * 8 + r;
                if (row >= M) continue;
                float v = acc[i][j][r] * alpha + bv;
                if (resid) v += resid[(long long)row * ldr + col];
                if (act)   v = 0.5f * v * (1.0f + erff(v * 0.70710678118654752f));
                if (Cf) Cf[(long long)row * ldcf + col] = v;
                if (Ch) {
                    if (transCh) Ch[(long long)col * ldch + row] = (half_t)v;
                    else         Ch[(long long)row * ldch + col] = (half_t)v;
                }
            }
        }
    }
}

// ---------------------------------------------------------------------------
// LayerNorm over 512 features, writes f16 (GEMM A-side).  grid(TSEQ, BATCH).
// ---------------------------------------------------------------------------
__global__ __launch_bounds__(128)
void ln_f16_kernel(const float* __restrict__ x, const float* __restrict__ g,
                   const float* __restrict__ bta, half_t* __restrict__ y)
{
    __shared__ float red[128];
    const long long row = (long long)blockIdx.y * MPAD + blockIdx.x;
    const float* xr = x + row * DMODEL;
    half_t*      yr = y + row * DMODEL;
    const int tid = threadIdx.x;

    float v0[4];
    float s = 0.0f;
    for (int k = 0; k < 4; ++k) { v0[k] = xr[tid + k * 128]; s += v0[k]; }
    red[tid] = s; __syncthreads();
    for (int off = 64; off > 0; off >>= 1) {
        if (tid < off) red[tid] += red[tid + off];
        __syncthreads();
    }
    const float mean = red[0] / (float)DMODEL;
    __syncthreads();
    s = 0.0f;
    for (int k = 0; k < 4; ++k) { float d = v0[k] - mean; s += d * d; }
    red[tid] = s; __syncthreads();
    for (int off = 64; off > 0; off >>= 1) {
        if (tid < off) red[tid] += red[tid + off];
        __syncthreads();
    }
    const float inv = rsqrtf(red[0] / (float)DMODEL + 1e-5f);
    for (int k = 0; k < 4; ++k) {
        const int c = tid + k * 128;
        yr[c] = (half_t)((v0[k] - mean) * inv * g[c] + bta[c]);
    }
}

// ---------------------------------------------------------------------------
// Row softmax over 513 scores; writes f16 probs, zeroing pad cols 513..543.
// grid(TSEQ, BATCH*NHEAD).  Row index = blockIdx.y*MPAD + blockIdx.x.
// ---------------------------------------------------------------------------
__global__ __launch_bounds__(256)
void softmax_kernel(const float* __restrict__ S, half_t* __restrict__ P)
{
    __shared__ float red[256];
    const long long row = (long long)blockIdx.y * MPAD + blockIdx.x;
    const float* sr = S + row * SPAD;
    half_t*      pr = P + row * SPAD;
    const int tid = threadIdx.x;

    float m = -3.4e38f;
    for (int i = tid; i < TSEQ; i += 256) m = fmaxf(m, sr[i]);
    red[tid] = m; __syncthreads();
    for (int off = 128; off > 0; off >>= 1) {
        if (tid < off) red[tid] = fmaxf(red[tid], red[tid + off]);
        __syncthreads();
    }
    m = red[0]; __syncthreads();

    float s = 0.0f;
    for (int i = tid; i < TSEQ; i += 256) s += __expf(sr[i] - m);
    red[tid] = s; __syncthreads();
    for (int off = 128; off > 0; off >>= 1) {
        if (tid < off) red[tid] += red[tid + off];
        __syncthreads();
    }
    const float inv = 1.0f / red[0];
    for (int i = tid; i < SPAD; i += 256)
        pr[i] = (i < TSEQ) ? (half_t)(__expf(sr[i] - m) * inv) : (half_t)0.0f;
}

// ---------------------------------------------------------------------------
// Small glue kernels
// ---------------------------------------------------------------------------
__global__ void f32_to_f16_kernel(const float* __restrict__ in, half_t* __restrict__ out,
                                  long long n)
{
    long long i = (long long)blockIdx.x * blockDim.x + threadIdx.x;
    const long long stride = (long long)gridDim.x * blockDim.x;
    for (; i < n; i += stride) out[i] = (half_t)in[i];
}

__global__ void zero_f16_kernel(half_t* __restrict__ p, long long n)
{
    long long i = (long long)blockIdx.x * blockDim.x + threadIdx.x;
    const long long stride = (long long)gridDim.x * blockDim.x;
    for (; i < n; i += stride) p[i] = (half_t)0.0f;
}

// out[b][0][:] = token ; out[b][1+t][:] = fin[b][t][:]   (rows padded to MPAD)
__global__ void build_seq_kernel(const float* __restrict__ fin, const float* __restrict__ token,
                                 float* __restrict__ out)
{
    const long long n = (long long)BATCH * TSEQ * DMODEL;
    long long i = (long long)blockIdx.x * blockDim.x + threadIdx.x;
    const long long stride = (long long)gridDim.x * blockDim.x;
    for (; i < n; i += stride) {
        const int       d = (int)(i % DMODEL);
        const long long t = (i / DMODEL) % TSEQ;
        const long long b = i / ((long long)TSEQ * DMODEL);
        out[(b * MPAD + t) * DMODEL + d] =
            (t == 0) ? token[d] : fin[(b * 512 + (t - 1)) * DMODEL + d];
    }
}

// MIX[b][0][:] = G row 0 ; MIX[b][t>=1][:] = L rows (stop_gradient == identity fwd)
__global__ void mix_build_kernel(const float* __restrict__ G, const float* __restrict__ L,
                                 float* __restrict__ MIX)
{
    const long long n = (long long)BATCH * TSEQ * DMODEL;
    long long i = (long long)blockIdx.x * blockDim.x + threadIdx.x;
    const long long stride = (long long)gridDim.x * blockDim.x;
    for (; i < n; i += stride) {
        const int       d = (int)(i % DMODEL);
        const long long t = (i / DMODEL) % TSEQ;
        const long long b = i / ((long long)TSEQ * DMODEL);
        const long long idx = (b * MPAD + t) * DMODEL + d;
        MIX[idx] = (t == 0) ? G[idx] : L[idx];
    }
}

// dst[b][0][:] = src[b][0][:]
__global__ void copy_row0_kernel(const float* __restrict__ src, float* __restrict__ dst)
{
    const long long n = (long long)BATCH * DMODEL;
    long long i = (long long)blockIdx.x * blockDim.x + threadIdx.x;
    const long long stride = (long long)gridDim.x * blockDim.x;
    for (; i < n; i += stride) {
        const long long b = i / DMODEL;
        const int       d = (int)(i % DMODEL);
        dst[(b * MPAD) * DMODEL + d] = src[(b * MPAD) * DMODEL + d];
    }
}

// strip MPAD row padding -> dense (B, TSEQ, DMODEL)
__global__ void copy_out_kernel(const float* __restrict__ src, float* __restrict__ dst)
{
    const long long n = (long long)BATCH * TSEQ * DMODEL;
    long long i = (long long)blockIdx.x * blockDim.x + threadIdx.x;
    const long long stride = (long long)gridDim.x * blockDim.x;
    for (; i < n; i += stride) {
        const int       d = (int)(i % DMODEL);
        const long long t = (i / DMODEL) % TSEQ;
        const long long b = i / ((long long)TSEQ * DMODEL);
        dst[i] = src[(b * MPAD + t) * DMODEL + d];
    }
}

// ---------------------------------------------------------------------------
// Host orchestration
// ---------------------------------------------------------------------------
namespace {

struct Scratch {
    float  *G, *L, *MIX, *X1, *S;
    half_t *XF16, *OF16, *QF, *KF, *VT, *H1, *P;
    half_t *Wq, *Wk, *Wv, *Wp, *W1, *W2;
};

inline void launch_gemm(hipStream_t st,
                        const half_t* A, long long sAh, long long sAb, int lda,
                        const half_t* B, long long sBh, long long sBb, int ldbn,
                        int M, int N, int K, int Z, int zdiv,
                        const float* bias,
                        const float* resid, long long sRh, long long sRb, int ldr,
                        float* Cf, long long sCfh, long long sCfb, int ldcf,
                        half_t* Ch, long long sChh, long long sChb, int ldch,
                        int transCh, float alpha, int act)
{
    dim3 grid((N + 63) / 64, (M + 127) / 128, Z);
    gemm_wmma_f16<<<grid, 256, 0, st>>>(A, sAh, sAb, lda, B, sBh, sBb, ldbn,
                                        M, N, K, zdiv, bias, resid, sRh, sRb, ldr,
                                        Cf, sCfh, sCfb, ldcf, Ch, sChh, sChb, ldch,
                                        transCh, alpha, act);
}

// One transformer block, in place on X ([BATCH][MPAD][DMODEL] fp32).
// prm order: 0 lnv_g,1 lnv_b,2 wq,3 wk,4 wv,5 wp,6 bp,7 ln2_g,8 ln2_b,9 w1,10 b1,11 w2,12 b2
void run_block(hipStream_t st, float* X, const float* const* prm, const Scratch& w)
{
    const long long WN = (long long)DMODEL * DMODEL;
    const long long WF = (long long)DMODEL * DFF;
    f32_to_f16_kernel<<<256, 256, 0, st>>>(prm[2],  w.Wq, WN);
    f32_to_f16_kernel<<<256, 256, 0, st>>>(prm[3],  w.Wk, WN);
    f32_to_f16_kernel<<<256, 256, 0, st>>>(prm[4],  w.Wv, WN);
    f32_to_f16_kernel<<<256, 256, 0, st>>>(prm[5],  w.Wp, WN);
    f32_to_f16_kernel<<<256, 256, 0, st>>>(prm[9],  w.W1, WF);
    f32_to_f16_kernel<<<256, 256, 0, st>>>(prm[11], w.W2, WF);

    const long long AB = (long long)MPAD * DMODEL;   // per-batch activation stride
    const long long SH = (long long)MPAD * SPAD;     // per-head score stride
    const long long HB = (long long)MPAD * DFF;
    const long long VB = (long long)DMODEL * SPAD;   // per-batch V^T stride

    // LN1 -> f16
    ln_f16_kernel<<<dim3(TSEQ, BATCH), 128, 0, st>>>(X, prm[0], prm[1], w.XF16);

    // Q (scaled), K : x @ w.T   (row-major f16)
    launch_gemm(st, w.XF16, 0, AB, DMODEL, w.Wq, 0, 0, DMODEL,
                TSEQ, DMODEL, DMODEL, BATCH, 1,
                nullptr, nullptr, 0, 0, 0, nullptr, 0, 0, 0,
                w.QF, 0, AB, DMODEL, 0, SCALE_Q, 0);
    launch_gemm(st, w.XF16, 0, AB, DMODEL, w.Wk, 0, 0, DMODEL,
                TSEQ, DMODEL, DMODEL, BATCH, 1,
                nullptr, nullptr, 0, 0, 0, nullptr, 0, 0, 0,
                w.KF, 0, AB, DMODEL, 0, 1.0f, 0);
    // V : written TRANSPOSED -> VT[b][h*64+d][t]  (k-contiguous B side for PV)
    launch_gemm(st, w.XF16, 0, AB, DMODEL, w.Wv, 0, 0, DMODEL,
                TSEQ, DMODEL, DMODEL, BATCH, 1,
                nullptr, nullptr, 0, 0, 0, nullptr, 0, 0, 0,
                w.VT, 0, VB, SPAD, 1, 1.0f, 0);

    // S[b,h] = Qh @ Kh^T   (per-head column offset h*64, lda=512)
    launch_gemm(st, w.QF, HDIM, AB, DMODEL, w.KF, HDIM, AB, DMODEL,
                TSEQ, TSEQ, HDIM, BATCH * NHEAD, NHEAD,
                nullptr, nullptr, 0, 0, 0,
                w.S, SH, (long long)NHEAD * SH, SPAD,
                nullptr, 0, 0, 0, 0, 1.0f, 0);

    softmax_kernel<<<dim3(TSEQ, BATCH * NHEAD), 256, 0, st>>>(w.S, w.P);

    // O[b,h] = P @ V : B = VT rows (h*64+d), k-contiguous over t
    launch_gemm(st, w.P, SH, (long long)NHEAD * SH, SPAD,
                w.VT, (long long)HDIM * SPAD, VB, SPAD,
                TSEQ, HDIM, SPAD, BATCH * NHEAD, NHEAD,
                nullptr, nullptr, 0, 0, 0, nullptr, 0, 0, 0,
                w.OF16, HDIM, AB, DMODEL, 0, 1.0f, 0);

    // x1 = O @ wp.T + bp + x
    launch_gemm(st, w.OF16, 0, AB, DMODEL, w.Wp, 0, 0, DMODEL,
                TSEQ, DMODEL, DMODEL, BATCH, 1,
                prm[6], X, 0, AB, DMODEL,
                w.X1, 0, AB, DMODEL, nullptr, 0, 0, 0, 0, 1.0f, 0);

    // LN2 -> f16
    ln_f16_kernel<<<dim3(TSEQ, BATCH), 128, 0, st>>>(w.X1, prm[7], prm[8], w.XF16);

    // h = gelu(x @ w1.T + b1)  (f16 out)
    launch_gemm(st, w.XF16, 0, AB, DMODEL, w.W1, 0, 0, DMODEL,
                TSEQ, DFF, DMODEL, BATCH, 1,
                prm[10], nullptr, 0, 0, 0, nullptr, 0, 0, 0,
                w.H1, 0, HB, DFF, 0, 1.0f, 1);

    // x = h @ w2.T + b2 + x1   (writes block output in place)
    launch_gemm(st, w.H1, 0, HB, DFF, w.W2, 0, 0, DFF,
                TSEQ, DMODEL, DFF, BATCH, 1,
                prm[12], w.X1, 0, AB, DMODEL,
                X, 0, AB, DMODEL, nullptr, 0, 0, 0, 0, 1.0f, 0);
}

} // namespace

extern "C" void kernel_launch(void* const* d_in, const int* in_sizes, int n_in,
                              void* d_out, int out_size, void* d_ws, size_t ws_size,
                              hipStream_t stream)
{
    (void)in_sizes; (void)n_in; (void)out_size; (void)ws_size;

    const float* f_g   = (const float*)d_in[0];
    const float* f_l   = (const float*)d_in[1];
    const float* tok_g = (const float*)d_in[2];
    const float* tok_l = (const float*)d_in[3];

    const float* enc[6][13];
    const float* msa[3][13];
    for (int i = 0; i < 6; ++i)
        for (int j = 0; j < 13; ++j) enc[i][j] = (const float*)d_in[4 + i * 13 + j];
    for (int m = 0; m < 3; ++m)
        for (int j = 0; j < 13; ++j) msa[m][j] = (const float*)d_in[4 + 78 + m * 13 + j];

    // carve workspace (256B aligned slabs)
    char*  base = (char*)d_ws;
    size_t off  = 0;
    auto carve = [&](size_t bytes) -> void* {
        void* p = base + off;
        off = (off + bytes + 255) & ~(size_t)255;
        return p;
    };
    const size_t act32 = (size_t)BATCH * MPAD * DMODEL * sizeof(float);
    const size_t act16 = (size_t)BATCH * MPAD * DMODEL * sizeof(half_t);
    const size_t sco32 = (size_t)BATCH * NHEAD * MPAD * SPAD * sizeof(float);
    const size_t sco16 = (size_t)BATCH * NHEAD * MPAD * SPAD * sizeof(half_t);

    Scratch w;
    w.G    = (float*)carve(act32);
    w.L    = (float*)carve(act32);
    w.MIX  = (float*)carve(act32);
    w.X1   = (float*)carve(act32);
    w.S    = (float*)carve(sco32);
    w.XF16 = (half_t*)carve(act16);
    w.OF16 = (half_t*)carve(act16);
    w.QF   = (half_t*)carve(act16);
    w.KF   = (half_t*)carve(act16);
    w.VT   = (half_t*)carve((size_t)BATCH * DMODEL * SPAD * sizeof(half_t));
    w.P    = (half_t*)carve(sco16);
    w.H1   = (half_t*)carve((size_t)BATCH * MPAD * DFF * sizeof(half_t));
    w.Wq   = (half_t*)carve((size_t)DMODEL * DMODEL * sizeof(half_t));
    w.Wk   = (half_t*)carve((size_t)DMODEL * DMODEL * sizeof(half_t));
    w.Wv   = (half_t*)carve((size_t)DMODEL * DMODEL * sizeof(half_t));
    w.Wp   = (half_t*)carve((size_t)DMODEL * DMODEL * sizeof(half_t));
    w.W1   = (half_t*)carve((size_t)DMODEL * DFF * sizeof(half_t));
    w.W2   = (half_t*)carve((size_t)DMODEL * DFF * sizeof(half_t));

    // V^T tail columns (t = 513..543) feed the padded K-reduction of PV;
    // P pad cols are zero, but 0 * NaN would poison -> zero V^T once per call.
    zero_f16_kernel<<<256, 256, 0, stream>>>(w.VT, (long long)BATCH * DMODEL * SPAD);

    // prepend tokens
    build_seq_kernel<<<512, 256, 0, stream>>>(f_g, tok_g, w.G);
    build_seq_kernel<<<512, 256, 0, stream>>>(f_l, tok_l, w.L);

    auto mix_fuse = [&](const float* const* p) {
        mix_build_kernel<<<512, 256, 0, stream>>>(w.G, w.L, w.MIX);
        run_block(stream, w.MIX, p, w);
        copy_row0_kernel<<<8, 128, 0, stream>>>(w.MIX, w.G);
    };

    mix_fuse(msa[0]);
    for (int i = 0; i < 6; ++i) {
        run_block(stream, w.G, enc[i], w);
        run_block(stream, w.L, enc[i], w);
        if (i == 3) mix_fuse(msa[1]);   // LAYERS // 2
    }
    mix_fuse(msa[2]);

    float* out = (float*)d_out;
    copy_out_kernel<<<512, 256, 0, stream>>>(w.G, out);
    copy_out_kernel<<<512, 256, 0, stream>>>(w.L, out + (long long)BATCH * TSEQ * DMODEL);
}
